// MLAttention_70360154243181
// MI455X (gfx1250) — matrix operand order repaired and checked by reading, not compiled, not run on hardware
//
#include <hip/hip_runtime.h>

typedef __bf16 bf16;
typedef __attribute__((ext_vector_type(16))) __bf16 v16bf;
typedef __attribute__((ext_vector_type(8)))  __bf16 v8bf;
typedef __attribute__((ext_vector_type(8)))  float  v8f;

#define B_  4
#define T_  2048
#define DM_ 2048
#define H_  16
#define D_  128
#define R_  512
#define BT_ (B_ * T_)

#define WMMA_BF16(a, b, c) \
  __builtin_amdgcn_wmma_f32_16x16x32_bf16(false, (a), false, (b), (short)0, (c), false, false)

// ---------- helpers ----------
__device__ __forceinline__ bf16 f2bf(float f) {
  union { float f; unsigned u; } a; a.f = f;
  unsigned r = a.u + 0x7fffu + ((a.u >> 16) & 1u);   // round-to-nearest-even
  unsigned short h = (unsigned short)(r >> 16);
  bf16 o; __builtin_memcpy(&o, &h, 2); return o;
}

// CDNA5 async global->LDS copy (16B per lane), tracked by ASYNCcnt.
__device__ __forceinline__ void async_load_b128(unsigned lds_off, const bf16* g) {
  asm volatile("global_load_async_to_lds_b128 %0, %1, off"
               :: "v"(lds_off), "v"((unsigned long long)(const void*)g)
               : "memory");
}
__device__ __forceinline__ void wait_async0() {
  asm volatile("s_wait_asynccnt 0x0" ::: "memory");
}
__device__ __forceinline__ unsigned lds_off_of(const void* p) {
  return (unsigned)(unsigned long long)p;   // low 32 bits of generic ptr = LDS offset
}

// Load a 16x32 bf16 WMMA fragment (A layout: lane = row%16, half splits K;
// elements 0..7 -> K = 8*half..8*half+7, elements 8..15 -> K = 16+8*half..).
// Same loader serves B when data is stored [N][K] (K-major per column).
__device__ __forceinline__ v16bf ld_frag(const bf16* base, int ld) {
  int lane = threadIdx.x & 31;
  int r = lane & 15, hf = lane >> 4;
  const bf16* p = base + r * ld + 8 * hf;
  v8bf lo = *(const v8bf*)(p);
  v8bf hi = *(const v8bf*)(p + 16);
  v16bf f;
#pragma unroll
  for (int i = 0; i < 8; i++) { f[i] = lo[i]; f[i + 8] = hi[i]; }
  return f;
}

// ---------- f32 -> bf16 conversion ----------
__global__ void cvt_bf16_kernel(const float* __restrict__ in, bf16* __restrict__ out, size_t n) {
  size_t i = (size_t)blockIdx.x * 256 + threadIdx.x;
  if (i < n) out[i] = f2bf(in[i]);
}

// ---------- bf16 GEMM: C(f32)[M,N] = A[M,K] @ B[K,N] ----------
// 128x128x32 macro-tile, 8 waves, each wave 32(M)x64(N) = 8 WMMA / k-step.
// A-tiles: async global->LDS, double-buffered; B-tiles: register transpose.
__global__ __launch_bounds__(256) void gemm_bf16_kernel(
    const bf16* __restrict__ A, const bf16* __restrict__ Bm, float* __restrict__ C,
    int M, int N, int K) {
  __shared__ bf16 As[2][128][32];
  __shared__ bf16 Bt[2][128][32];   // [n][k]
  int tid = threadIdx.x;
  int wave = tid >> 5, lane = tid & 31;
  int m0 = blockIdx.y * 128, n0 = blockIdx.x * 128;
  int wm = wave >> 1, wn = wave & 1;     // 4x2 wave grid: 32(M) x 64(N) per wave
  v8f acc[2][4];
#pragma unroll
  for (int mi = 0; mi < 2; mi++)
#pragma unroll
    for (int ni = 0; ni < 4; ni++) acc[mi][ni] = (v8f)(0.0f);

  int ar = tid >> 1, ac = (tid & 1) * 16;          // A tile cover: 128 rows x 32 k
  const bf16* Ag = A + (size_t)(m0 + ar) * K + ac;
  unsigned lA0 = lds_off_of(&As[0][ar][ac]);
  unsigned lA1 = lds_off_of(&As[1][ar][ac]);

  int bk = tid >> 3, bn = (tid & 7) * 16;          // B tile cover: 32 k x 128 n
  const bf16* Bg = Bm + (size_t)bk * N + n0 + bn;

  int nk = K >> 5;
  // preload tile 0
  async_load_b128(lA0, Ag);
  async_load_b128(lA0 + 16, Ag + 8);
  v8bf br0 = *(const v8bf*)(Bg);
  v8bf br1 = *(const v8bf*)(Bg + 8);

  for (int kt = 0; kt < nk; kt++) {
    int buf = kt & 1;
    // stage B(kt) into Bt[buf] (transpose); disjoint from Bt[buf^1] still being read
#pragma unroll
    for (int i = 0; i < 8; i++) {
      Bt[buf][bn + i][bk]     = br0[i];
      Bt[buf][bn + 8 + i][bk] = br1[i];
    }
    if (kt + 1 < nk) {                     // prefetch B(kt+1) into registers
      const bf16* Bg2 = Bg + (size_t)(kt + 1) * 32 * N;
      br0 = *(const v8bf*)(Bg2);
      br1 = *(const v8bf*)(Bg2 + 8);
    }
    wait_async0();                         // A(kt) landed in As[buf]
    __syncthreads();                       // tile(kt) fully staged by all waves
    if (kt + 1 < nk) {                     // async prefetch A(kt+1) into As[buf^1]
      const bf16* Ag2 = Ag + (kt + 1) * 32;
      async_load_b128(buf ? lA0 : lA1, Ag2);
      async_load_b128((buf ? lA0 : lA1) + 16, Ag2 + 8);
    }
    v16bf af0 = ld_frag(&As[buf][wm * 32][0], 32);
    v16bf af1 = ld_frag(&As[buf][wm * 32 + 16][0], 32);
#pragma unroll
    for (int ni = 0; ni < 4; ni++) {
      v16bf bfv = ld_frag(&Bt[buf][wn * 64 + ni * 16][0], 32);
      acc[0][ni] = WMMA_BF16(af0, bfv, acc[0][ni]);
      acc[1][ni] = WMMA_BF16(af1, bfv, acc[1][ni]);
    }
  }

  int cn = lane & 15, hf = lane >> 4;
#pragma unroll
  for (int mi = 0; mi < 2; mi++)
#pragma unroll
    for (int j = 0; j < 8; j++) {
      size_t row = (size_t)(m0 + wm * 32 + mi * 16 + j + 8 * hf);
      float* crow = C + row * N + n0 + wn * 64 + cn;
#pragma unroll
      for (int ni = 0; ni < 4; ni++) crow[ni * 16] = acc[mi][ni][j];
    }
}

// ---------- RoPE on q (with 1/sqrt(D) scale), transpose to [B,H,T,D] bf16 ----------
__global__ void rope_q_kernel(const float* __restrict__ q, const float* __restrict__ cs,
                              const float* __restrict__ sn, bf16* __restrict__ qo) {
  size_t idx = (size_t)blockIdx.x * 256 + threadIdx.x;   // B*T*H*64 threads
  int i = (int)(idx & 63);
  size_t t1 = idx >> 6;
  int h = (int)(t1 & (H_ - 1));
  size_t t2 = t1 >> 4;
  int t = (int)(t2 & (T_ - 1));
  int b = (int)(t2 >> 11);
  float c = cs[t * 64 + i], s = sn[t * 64 + i];
  size_t src = (((size_t)(b * T_ + t) * H_ + h) * D_) + 2 * i;
  float x0 = q[src], x1 = q[src + 1];
  const float rs = 0.088388347648318447f;   // 1/sqrt(128)
  size_t dst = (((size_t)(b * H_ + h) * T_ + t) * D_) + 2 * i;
  qo[dst]     = f2bf((x0 * c - x1 * s) * rs);
  qo[dst + 1] = f2bf((x1 * c + x0 * s) * rs);
}

// ---------- RoPE on k, passthrough v; kv f32 [B,T,2,H,D] -> k,v bf16 [B,H,T,D] ----------
__global__ void rope_kv_kernel(const float* __restrict__ kv, const float* __restrict__ cs,
                               const float* __restrict__ sn, bf16* __restrict__ ko,
                               bf16* __restrict__ vo) {
  size_t idx = (size_t)blockIdx.x * 256 + threadIdx.x;
  int i = (int)(idx & 63);
  size_t t1 = idx >> 6;
  int h = (int)(t1 & (H_ - 1));
  size_t t2 = t1 >> 4;
  int t = (int)(t2 & (T_ - 1));
  int b = (int)(t2 >> 11);
  float c = cs[t * 64 + i], s = sn[t * 64 + i];
  size_t base = (size_t)(b * T_ + t) * (2 * H_ * D_);
  size_t ks = base + (size_t)h * D_ + 2 * i;
  size_t vs = base + (size_t)H_ * D_ + (size_t)h * D_ + 2 * i;
  float k0 = kv[ks], k1 = kv[ks + 1];
  size_t dst = (((size_t)(b * H_ + h) * T_ + t) * D_) + 2 * i;
  ko[dst]     = f2bf(k0 * c - k1 * s);
  ko[dst + 1] = f2bf(k1 * c + k0 * s);
  vo[dst]     = f2bf(kv[vs]);
  vo[dst + 1] = f2bf(kv[vs + 1]);
}

// ---------- causal flash attention; q,k,v bf16 [B,H,T,D]; out bf16 [B,T,H*D] ----------
__global__ __launch_bounds__(256) void mla_attn_kernel(
    const bf16* __restrict__ Q, const bf16* __restrict__ Kk,
    const bf16* __restrict__ Vv, bf16* __restrict__ Oo) {
  __shared__ bf16 Kt[32][128];       // [key][d]  (async-filled)
  __shared__ bf16 Vt[128][32];       // [d][key]  (transposed)
  __shared__ bf16 Ps[8][16][32];     // per-wave P scratch [row][key]
  int tid = threadIdx.x, wave = tid >> 5, lane = tid & 31;
  int r = lane & 15, hf = lane >> 4;
  int bh = blockIdx.y;               // b*H + h
  int q0 = blockIdx.x * 128 + wave * 16;
  const bf16* Qb = Q  + (size_t)bh * T_ * D_;
  const bf16* Kb = Kk + (size_t)bh * T_ * D_;
  const bf16* Vb = Vv + (size_t)bh * T_ * D_;

  v16bf qf[4];
#pragma unroll
  for (int c = 0; c < 4; c++) qf[c] = ld_frag(Qb + (size_t)q0 * D_ + c * 32, D_);

  float mrun[8], lrun[8];
  v8f o[8];
#pragma unroll
  for (int j = 0; j < 8; j++) { mrun[j] = -__builtin_inff(); lrun[j] = 0.f; }
#pragma unroll
  for (int ds = 0; ds < 8; ds++) o[ds] = (v8f)(0.0f);

  int nkt = blockIdx.x * 4 + 4;      // (block_last_query+1)/32
  int lr = tid >> 3;                 // 0..31 key row
  int lc = (tid & 7) * 16;           // d chunk
  unsigned lK = lds_off_of(&Kt[lr][lc]);
  for (int kt = 0; kt < nkt; kt++) {
    int krow = kt * 32 + lr;
    const bf16* kg = Kb + (size_t)krow * D_ + lc;
    v8bf v0v = *(const v8bf*)(Vb + (size_t)krow * D_ + lc);
    v8bf v1v = *(const v8bf*)(Vb + (size_t)krow * D_ + lc + 8);
    __syncthreads();                 // previous tile's readers done
    async_load_b128(lK, kg);
    async_load_b128(lK + 16, kg + 8);
#pragma unroll
    for (int i = 0; i < 8; i++) { Vt[lc + i][lr] = v0v[i]; Vt[lc + 8 + i][lr] = v1v[i]; }
    wait_async0();
    __syncthreads();

    v8f s[2];
#pragma unroll
    for (int sub = 0; sub < 2; sub++) {
      v8f sc = (v8f)(0.0f);
#pragma unroll
      for (int c = 0; c < 4; c++) {
        v16bf kf = ld_frag(&Kt[sub * 16][c * 32], 128);
        sc = WMMA_BF16(qf[c], kf, sc);
      }
      int keyg = kt * 32 + sub * 16 + r;
#pragma unroll
      for (int j = 0; j < 8; j++) {
        int qg = q0 + j + 8 * hf;
        if (keyg > qg) sc[j] = -__builtin_inff();
      }
      s[sub] = sc;
    }

#pragma unroll
    for (int j = 0; j < 8; j++) {
      float rm = fmaxf(s[0][j], s[1][j]);
#pragma unroll
      for (int m = 8; m >= 1; m >>= 1) rm = fmaxf(rm, __shfl_xor(rm, m, 32));
      float mn = fmaxf(mrun[j], rm);
      bool dead = (mn == -__builtin_inff());
      float scl = dead ? 1.0f : __expf(mrun[j] - mn);
      float p0  = dead ? 0.0f : __expf(s[0][j] - mn);
      float p1  = dead ? 0.0f : __expf(s[1][j] - mn);
      float rsum = p0 + p1;
#pragma unroll
      for (int m = 8; m >= 1; m >>= 1) rsum += __shfl_xor(rsum, m, 32);
      lrun[j] = lrun[j] * scl + rsum;
      mrun[j] = mn;
#pragma unroll
      for (int ds = 0; ds < 8; ds++) o[ds][j] *= scl;
      int row = j + 8 * hf;
      Ps[wave][row][r]      = f2bf(p0);
      Ps[wave][row][16 + r] = f2bf(p1);
    }
    asm volatile("s_wait_dscnt 0x0" ::: "memory");
    v16bf pf = ld_frag(&Ps[wave][0][0], 32);
#pragma unroll
    for (int ds = 0; ds < 8; ds++) {
      v16bf vf = ld_frag(&Vt[ds * 16][0], 32);
      o[ds] = WMMA_BF16(pf, vf, o[ds]);
    }
  }

  int b = bh >> 4, h = bh & 15;
#pragma unroll
  for (int j = 0; j < 8; j++) {
    float inv = (lrun[j] > 0.f) ? 1.0f / lrun[j] : 0.f;
    int qg = q0 + j + 8 * hf;
    bf16* dst = Oo + (size_t)(b * T_ + qg) * (H_ * D_) + (size_t)h * D_;
#pragma unroll
    for (int ds = 0; ds < 8; ds++) dst[ds * 16 + r] = f2bf(o[ds][j] * inv);
  }
}

// ---------- launcher ----------
extern "C" void kernel_launch(void* const* d_in, const int* in_sizes, int n_in,
                              void* d_out, int out_size, void* d_ws, size_t ws_size,
                              hipStream_t stream) {
  const float* x    = (const float*)d_in[0];
  const float* cosb = (const float*)d_in[1];
  const float* sinb = (const float*)d_in[2];
  const float* Wq   = (const float*)d_in[3];
  const float* Wkd  = (const float*)d_in[4];
  const float* Wku  = (const float*)d_in[5];
  const float* Wo   = (const float*)d_in[6];
  float* out = (float*)d_out;

  char* ws = (char*)d_ws;
  size_t off = 0;
  auto alloc = [&](size_t bytes) {
    void* p = ws + off;
    off = (off + bytes + 255) & ~(size_t)255;
    return p;
  };
  bf16*  x_bf     = (bf16*)alloc((size_t)BT_ * DM_ * 2);
  bf16*  Wq_bf    = (bf16*)alloc((size_t)DM_ * (H_ * D_) * 2);
  bf16*  Wkd_bf   = (bf16*)alloc((size_t)DM_ * R_ * 2);
  bf16*  Wku_bf   = (bf16*)alloc((size_t)R_ * (2 * H_ * D_) * 2);
  bf16*  Wo_bf    = (bf16*)alloc((size_t)(H_ * D_) * DM_ * 2);
  float* qf32     = (float*)alloc((size_t)BT_ * (H_ * D_) * 4);
  float* kvlat_f  = (float*)alloc((size_t)BT_ * R_ * 4);
  bf16*  kvlat_bf = (bf16*)alloc((size_t)BT_ * R_ * 2);
  float* kv_f     = (float*)alloc((size_t)BT_ * (2 * H_ * D_) * 4);
  bf16*  q_bf     = (bf16*)alloc((size_t)BT_ * (H_ * D_) * 2);
  bf16*  k_bf     = (bf16*)alloc((size_t)BT_ * (H_ * D_) * 2);
  bf16*  v_bf     = (bf16*)alloc((size_t)BT_ * (H_ * D_) * 2);
  bf16*  attn_bf  = (bf16*)alloc((size_t)BT_ * (H_ * D_) * 2);

  auto cvt = [&](const float* src, bf16* dst, size_t n) {
    cvt_bf16_kernel<<<(unsigned)((n + 255) / 256), 256, 0, stream>>>(src, dst, n);
  };
  cvt(x,   x_bf,   (size_t)BT_ * DM_);
  cvt(Wq,  Wq_bf,  (size_t)DM_ * (H_ * D_));
  cvt(Wkd, Wkd_bf, (size_t)DM_ * R_);
  cvt(Wku, Wku_bf, (size_t)R_ * (2 * H_ * D_));
  cvt(Wo,  Wo_bf,  (size_t)(H_ * D_) * DM_);

  // kv_lat = x @ Wkd    [8192,2048]x[2048,512]
  gemm_bf16_kernel<<<dim3(R_ / 128, BT_ / 128), 256, 0, stream>>>(
      x_bf, Wkd_bf, kvlat_f, BT_, R_, DM_);
  cvt(kvlat_f, kvlat_bf, (size_t)BT_ * R_);
  // kv = kv_lat @ Wku   [8192,512]x[512,4096]
  gemm_bf16_kernel<<<dim3((2 * H_ * D_) / 128, BT_ / 128), 256, 0, stream>>>(
      kvlat_bf, Wku_bf, kv_f, BT_, 2 * H_ * D_, R_);
  // q = x @ Wq          [8192,2048]x[2048,2048]
  gemm_bf16_kernel<<<dim3((H_ * D_) / 128, BT_ / 128), 256, 0, stream>>>(
      x_bf, Wq_bf, qf32, BT_, H_ * D_, DM_);

  size_t nrope = (size_t)B_ * T_ * H_ * (D_ / 2);
  rope_q_kernel<<<(unsigned)(nrope / 256), 256, 0, stream>>>(qf32, cosb, sinb, q_bf);
  rope_kv_kernel<<<(unsigned)(nrope / 256), 256, 0, stream>>>(kv_f, cosb, sinb, k_bf, v_bf);

  mla_attn_kernel<<<dim3(T_ / 128, B_ * H_), 256, 0, stream>>>(q_bf, k_bf, v_bf, attn_bf);

  // out = attn @ Wo     [8192,2048]x[2048,2048]
  gemm_bf16_kernel<<<dim3(DM_ / 128, BT_ / 128), 256, 0, stream>>>(
      attn_bf, Wo_bf, out, BT_, DM_, H_ * D_);
}
